// GCNEmbed_43559558316064
// MI455X (gfx1250) — compile-verified, matrix operand order checked by reading, not run on hardware
//
#include <hip/hip_runtime.h>

// GCN 2-layer forward for MI455X (gfx1250, wave32).
// GEMMs via v_wmma_f32_16x16x32_bf16 (f32 accumulate); edge gather/scatter is
// L2-resident (h = 51.2MB < 192MB L2) using hardware global_atomic_add_f32.

typedef __attribute__((ext_vector_type(16))) __bf16 v16bf;
typedef __attribute__((ext_vector_type(8)))  __bf16 v8bf;
typedef __attribute__((ext_vector_type(8)))  float  v8f;

#define DHID 128

static __device__ __forceinline__ void atomic_fadd(float* p, float v) {
  __hip_atomic_fetch_add(p, v, __ATOMIC_RELAXED, __HIP_MEMORY_SCOPE_AGENT);
}

// ---------------- small prep kernels ----------------

__global__ void f32_to_bf16_kernel(const float* __restrict__ s, __bf16* __restrict__ d, int n) {
  int i = blockIdx.x * blockDim.x + threadIdx.x;
  if (i < n) d[i] = (__bf16)s[i];
}

__global__ void deg_count_kernel(const int* __restrict__ dst, float* __restrict__ cnt, int E) {
  int e = blockIdx.x * blockDim.x + threadIdx.x;
  if (e < E) atomic_fadd(&cnt[dst[e]], 1.0f);
}

__global__ void deg_finalize_kernel(float* __restrict__ cnt, int n) {
  int i = blockIdx.x * blockDim.x + threadIdx.x;
  if (i < n) cnt[i] = rsqrtf(cnt[i] + 1.0f);  // deg includes self-loop -> always >= 1
}

// ---------------- WMMA GEMM: C[N,128] = A[N,128] @ W^T (W row-major [128,128] bf16) ----
// One wave per 16-row tile. A kept in registers as 4 K-tiles (K=128 = 4x32).
// Safe for in-place A==C (A fully loaded into registers before any store).

__global__ void gemm128_bf16_kernel(const float* A, const __bf16* __restrict__ Wb,
                                    float* C, int nRows) {
  const int lane = threadIdx.x & 31;
  const int wave = threadIdx.x >> 5;
  const int tile = blockIdx.x * (blockDim.x >> 5) + wave;
  const int row0 = tile * 16;
  if (row0 >= nRows) return;               // wave-uniform exit: EXEC stays all-1s for WMMA
  const int half = lane >> 4;              // K-half selector per ISA 16-bit A layout
  const int l15  = lane & 15;

  // Load + convert 16x128 f32 A tile -> 4 x v16bf (ISA 7.12.2 16-bit A 16x32 layout):
  // lane holds row l15; element e -> K = k0 + 8*half + (e&7) + (e<8 ? 0 : 16)
  v16bf a[4];
  const float* arow = A + (size_t)(row0 + l15) * DHID;
  #pragma unroll
  for (int kk = 0; kk < 4; ++kk) {
    const float* p = arow + kk * 32 + half * 8;
    float4 f0 = *(const float4*)(p);       // e = 0..3
    float4 f1 = *(const float4*)(p + 4);   // e = 4..7
    float4 f2 = *(const float4*)(p + 16);  // e = 8..11
    float4 f3 = *(const float4*)(p + 20);  // e = 12..15
    v16bf t;
    t[0]=(__bf16)f0.x; t[1]=(__bf16)f0.y; t[2]=(__bf16)f0.z;  t[3]=(__bf16)f0.w;
    t[4]=(__bf16)f1.x; t[5]=(__bf16)f1.y; t[6]=(__bf16)f1.z;  t[7]=(__bf16)f1.w;
    t[8]=(__bf16)f2.x; t[9]=(__bf16)f2.y; t[10]=(__bf16)f2.z; t[11]=(__bf16)f2.w;
    t[12]=(__bf16)f3.x;t[13]=(__bf16)f3.y;t[14]=(__bf16)f3.z; t[15]=(__bf16)f3.w;
    a[kk] = t;
  }

  #pragma unroll
  for (int j = 0; j < 8; ++j) {            // 8 column tiles of 16
    v8f acc = {};
    #pragma unroll
    for (int kk = 0; kk < 4; ++kk) {
      // B[k, n] = W[n, k]: lane n reads row (j*16 + l15) of W, same K shuffle as A.
      const __bf16* bp = Wb + (size_t)(j * 16 + l15) * DHID + kk * 32 + half * 8;
      v8bf blo = *(const v8bf*)(bp);       // e = 0..7
      v8bf bhi = *(const v8bf*)(bp + 16);  // e = 8..15
      v16bf b;
      #pragma unroll
      for (int i = 0; i < 8; ++i) { b[i] = blo[i]; b[8 + i] = bhi[i]; }
      acc = __builtin_amdgcn_wmma_f32_16x16x32_bf16(
          /*neg_a=*/false, a[kk], /*neg_b=*/false, b,
          /*c_mod=*/(short)0, acc, /*reuse_a=*/false, /*reuse_b=*/false);
    }
    // D layout: acc[v] = D[M = v + 8*half, N = l15]
    float* crow = C + (size_t)(row0 + half * 8) * DHID + j * 16 + l15;
    #pragma unroll
    for (int v = 0; v < 8; ++v) crow[(size_t)v * DHID] = acc[v];
  }
}

// ---------------- edge aggregation: agg[dst] += h[src] * dinv[src]*dinv[dst] ----------
// One wave per edge: 32 lanes x float4 = 128 channels; gathers hit L2 (h fits in L2).

__global__ void aggregate_kernel(const int* __restrict__ src, const int* __restrict__ dst,
                                 const float* __restrict__ dinv, const float* __restrict__ h,
                                 float* __restrict__ agg, int E) {
  const int lane = threadIdx.x & 31;
  const int wid  = (blockIdx.x * blockDim.x + threadIdx.x) >> 5;
  if (wid >= E) return;
  const int s = src[wid], d = dst[wid];
  const float nrm = dinv[s] * dinv[d];
  float4 v = ((const float4*)(h + (size_t)s * DHID))[lane];
  float* op = agg + (size_t)d * DHID + lane * 4;
  atomic_fadd(op + 0, v.x * nrm);
  atomic_fadd(op + 1, v.y * nrm);
  atomic_fadd(op + 2, v.z * nrm);
  atomic_fadd(op + 3, v.w * nrm);
}

// ---------------- fused epilogues (self-loop term folded in, no atomics) -------------

// h <- relu(agg + h*dinv^2 + b1)   (in-place on h)
__global__ void post1_kernel(float* __restrict__ h, const float* __restrict__ agg,
                             const float* __restrict__ dinv, const float* __restrict__ bias,
                             int n) {
  int idx = blockIdx.x * blockDim.x + threadIdx.x;   // n*32 float4 slots
  if (idx >= n * 32) return;
  int node = idx >> 5, c4 = idx & 31;
  float w = dinv[node]; w *= w;
  float4 hv = ((const float4*)h)[(size_t)node * 32 + c4];
  float4 av = ((const float4*)agg)[(size_t)node * 32 + c4];
  float4 bv = ((const float4*)bias)[c4];
  float4 r;
  r.x = fmaxf(av.x + hv.x * w + bv.x, 0.0f);
  r.y = fmaxf(av.y + hv.y * w + bv.y, 0.0f);
  r.z = fmaxf(av.z + hv.z * w + bv.z, 0.0f);
  r.w = fmaxf(av.w + hv.w * w + bv.w, 0.0f);
  ((float4*)h)[(size_t)node * 32 + c4] = r;
}

// out <- agg + h*dinv^2 + b2 + x   (residual)
__global__ void post2_kernel(const float* __restrict__ agg, const float* __restrict__ h,
                             const float* __restrict__ dinv, const float* __restrict__ bias,
                             const float* __restrict__ x, float* __restrict__ out, int n) {
  int idx = blockIdx.x * blockDim.x + threadIdx.x;
  if (idx >= n * 32) return;
  int node = idx >> 5, c4 = idx & 31;
  float w = dinv[node]; w *= w;
  float4 hv = ((const float4*)h)[(size_t)node * 32 + c4];
  float4 av = ((const float4*)agg)[(size_t)node * 32 + c4];
  float4 xv = ((const float4*)x)[(size_t)node * 32 + c4];
  float4 bv = ((const float4*)bias)[c4];
  float4 r;
  r.x = av.x + hv.x * w + bv.x + xv.x;
  r.y = av.y + hv.y * w + bv.y + xv.y;
  r.z = av.z + hv.z * w + bv.z + xv.z;
  r.w = av.w + hv.w * w + bv.w + xv.w;
  ((float4*)out)[(size_t)node * 32 + c4] = r;
}

// ---------------- driver ----------------

extern "C" void kernel_launch(void* const* d_in, const int* in_sizes, int n_in,
                              void* d_out, int out_size, void* d_ws, size_t ws_size,
                              hipStream_t stream) {
  (void)n_in; (void)out_size; (void)ws_size;
  const float* x  = (const float*)d_in[0];
  const int*   ei = (const int*)d_in[1];     // edge_index [2, E]
  const float* W1 = (const float*)d_in[2];
  const float* b1 = (const float*)d_in[3];
  const float* W2 = (const float*)d_in[4];
  const float* b2 = (const float*)d_in[5];
  float* out = (float*)d_out;

  const int N = in_sizes[0] / DHID;          // 100000
  const int E = in_sizes[1] / 2;             // 1600000
  const int* srcIdx = ei;
  const int* dstIdx = ei + E;

  // workspace layout (16B aligned throughout)
  float*  bufA = (float*)d_ws;               // N*128 f32 : h / h1 / h2
  float*  bufB = bufA + (size_t)N * DHID;    // N*128 f32 : aggregation target
  float*  dinv = bufB + (size_t)N * DHID;    // N f32     : deg count then rsqrt
  __bf16* W1b  = (__bf16*)(dinv + N);        // 128*128 bf16
  __bf16* W2b  = W1b + DHID * DHID;          // 128*128 bf16

  hipMemsetAsync(dinv, 0, (size_t)N * sizeof(float), stream);
  hipMemsetAsync(bufB, 0, (size_t)N * DHID * sizeof(float), stream);

  f32_to_bf16_kernel<<<(DHID * DHID + 255) / 256, 256, 0, stream>>>(W1, W1b, DHID * DHID);
  f32_to_bf16_kernel<<<(DHID * DHID + 255) / 256, 256, 0, stream>>>(W2, W2b, DHID * DHID);

  deg_count_kernel<<<(E + 255) / 256, 256, 0, stream>>>(dstIdx, dinv, E);
  deg_finalize_kernel<<<(N + 255) / 256, 256, 0, stream>>>(dinv, N);

  const int tiles  = (N + 15) / 16;          // 6250 (N is a multiple of 16)
  const int gblk   = (tiles + 7) / 8;        // 8 waves / 256-thread block
  const int ablk   = (E + 7) / 8;            // 1 wave per edge
  const int pblk   = (N * 32 + 255) / 256;

  // Layer 1
  gemm128_bf16_kernel<<<gblk, 256, 0, stream>>>(x, W1b, bufA, N);
  aggregate_kernel<<<ablk, 256, 0, stream>>>(srcIdx, dstIdx, dinv, bufA, bufB, E);
  post1_kernel<<<pblk, 256, 0, stream>>>(bufA, bufB, dinv, b1, N);
  hipMemsetAsync(bufB, 0, (size_t)N * DHID * sizeof(float), stream);

  // Layer 2 (GEMM in-place: each wave loads its 16 rows to registers before storing)
  gemm128_bf16_kernel<<<gblk, 256, 0, stream>>>(bufA, W2b, bufA, N);
  aggregate_kernel<<<ablk, 256, 0, stream>>>(srcIdx, dstIdx, dinv, bufA, bufB, E);
  post2_kernel<<<pblk, 256, 0, stream>>>(bufB, bufA, dinv, b2, x, out, N);
}
